// RCNN_ALEXNET_43705587204146
// MI455X (gfx1250) — compile-verified, matrix operand order checked by reading, not run on hardware
//
#include <hip/hip_runtime.h>

// ---------------------------------------------------------------------------
// Types for WMMA
// ---------------------------------------------------------------------------
typedef __attribute__((ext_vector_type(16))) _Float16 v16h;
typedef __attribute__((ext_vector_type(8)))  _Float16 v8h;
typedef __attribute__((ext_vector_type(8)))  float    v8f;

#define BN 128
#define BK 64              // K-depth per pipeline stage (2 wmma K-steps)
#define LDAH 72            // BK + 8 halves padding (144B LDS row stride)
#define ROUT 6

// gfx1250 async global->LDS copy, 16 bytes per lane, tracked by ASYNCcnt.
// lds_off = wave-relative LDS byte offset (low 32 bits of the generic pointer).
__device__ __forceinline__ void async_ld_b128(unsigned lds_off, const void* gptr) {
  asm volatile("global_load_async_to_lds_b128 %0, %1, off"
               :: "v"(lds_off), "v"(gptr)
               : "memory");
}

// ---------------------------------------------------------------------------
// Tiled f16 WMMA GEMM with 2-stage async-LDS pipeline (peeled epilogue).
//   C[M,N] = A[Mpad,Kpad] * Bw[Npad,Kpad]^T + bias, optional ReLU
// A: [Mpad x Kpad] f16 row-major, zero padded.  Bw: [Npad x Kpad] f16 row-major
// (= weight [N,K]).  256 threads = 8 waves.  Block tile TM x 128.
//   TM=128: wave tile 64x32 (4x2 frags, 16 wmma/stage)  -> conv GEMMs
//   TM=64 : wave tile 32x32 (2x2 frags,  8 wmma/stage)  -> FC GEMMs
// Kpad must be a multiple of 64 (true for every GEMM in this net).
// ---------------------------------------------------------------------------
template <int TM>
__global__ __launch_bounds__(256)
void gemm_f16_wmma(const _Float16* __restrict__ A,
                   const _Float16* __restrict__ Bw,
                   const float*    __restrict__ bias,
                   float*          __restrict__ C,
                   int M, int N, int Kpad, int relu)
{
  constexpr int STAGEH = (TM + BN) * LDAH;            // halves per stage
  constexpr int NCA    = (TM * BK) / (256 * 8);       // b128 copies per thread (A)
  constexpr int NCB    = (BN * BK) / (256 * 8);       // b128 copies per thread (B)
  constexpr int NASYNC = NCA + NCB;                   // async issues per wave/stage
  constexpr int MI     = TM / 32;                     // wmma frags along M per wave

  __shared__ __align__(16) _Float16 smem[2 * STAGEH];

  const int tid  = threadIdx.x;
  const int lane = tid & 31;
  const int wave = tid >> 5;
  const int wm   = wave >> 2;     // 0..1 -> TM/2-row slab
  const int wn   = wave & 3;      // 0..3 -> 32-col slab

  const long bm0 = (long)blockIdx.y * TM;
  const long bn0 = (long)blockIdx.x * BN;

  const unsigned smem_base = (unsigned)(size_t)&smem[0];

  v8f acc[MI][2];
#pragma unroll
  for (int i = 0; i < MI; ++i)
#pragma unroll
    for (int j = 0; j < 2; ++j)
#pragma unroll
      for (int r = 0; r < 8; ++r)
        acc[i][j][r] = 0.0f;

  const int frow  = lane & 15;               // M (or N) index inside a fragment
  const int kbase = (lane < 16) ? 0 : 8;     // per-lane K base per ISA layout

  // Issue one stage of async copies (A tile then B tile), 16B chunks.
  auto issue_stage = [&](int st, int k0) {
    const unsigned base = smem_base + (unsigned)(st * STAGEH * 2);
#pragma unroll
    for (int i = 0; i < NCA; ++i) {
      const int ch  = tid + i * 256;           // b128 chunk id over A tile
      const int row = ch >> 3, colh = (ch & 7) * 8;
      async_ld_b128(base + (unsigned)((row * LDAH + colh) * 2),
                    A + (bm0 + row) * (long)Kpad + k0 + colh);
    }
#pragma unroll
    for (int i = 0; i < NCB; ++i) {
      const int ch  = tid + i * 256;           // b128 chunk id over B tile
      const int row = ch >> 3, colh = (ch & 7) * 8;
      async_ld_b128(base + (unsigned)(((TM + row) * LDAH + colh) * 2),
                    Bw + (bn0 + row) * (long)Kpad + k0 + colh);
    }
  };

  // Consume one LDS stage: 2 wmma K-steps of 32.
  auto compute_stage = [&](int st) {
    const _Float16* sA = &smem[st * STAGEH];
    const _Float16* sB = &smem[st * STAGEH + TM * LDAH];
#pragma unroll
    for (int kk = 0; kk < BK; kk += 32) {
      v16h af[MI];
      v16h bf[2];
#pragma unroll
      for (int i = 0; i < MI; ++i) {
        const int mrow = wm * (TM / 2) + i * 16 + frow;
        const v8h lo = *(const v8h*)&sA[mrow * LDAH + kk + kbase];
        const v8h hi = *(const v8h*)&sA[mrow * LDAH + kk + kbase + 16];
#pragma unroll
        for (int e = 0; e < 8; ++e) { af[i][e] = lo[e]; af[i][e + 8] = hi[e]; }
      }
#pragma unroll
      for (int j = 0; j < 2; ++j) {
        const int nrow = wn * 32 + j * 16 + frow;
        const v8h lo = *(const v8h*)&sB[nrow * LDAH + kk + kbase];
        const v8h hi = *(const v8h*)&sB[nrow * LDAH + kk + kbase + 16];
#pragma unroll
        for (int e = 0; e < 8; ++e) { bf[j][e] = lo[e]; bf[j][e + 8] = hi[e]; }
      }
#pragma unroll
      for (int i = 0; i < MI; ++i)
#pragma unroll
        for (int j = 0; j < 2; ++j)
          acc[i][j] = __builtin_amdgcn_wmma_f32_16x16x32_f16(
              /*neg_a=*/false, af[i], /*neg_b=*/false, bf[j],
              /*c_mod=*/(short)0, acc[i][j], /*reuse_a=*/false, /*reuse_b=*/false);
    }
  };

  const int nsteps = Kpad / BK;
  issue_stage(0, 0);                           // prologue: stage 0 in flight

  // Steady state: branch-free issue->wait->compute pipeline.
  int s = 0;
  for (; s < nsteps - 1; ++s) {
    __syncthreads();                           // WAR: buf (s+1)&1 free everywhere
    issue_stage((s + 1) & 1, (s + 1) * BK);    // overlap next copy with compute
    if (NASYNC == 8)      asm volatile("s_wait_asynccnt 0x8" ::: "memory");
    else                  asm volatile("s_wait_asynccnt 0x6" ::: "memory");
    __syncthreads();                           // stage s visible to all waves
    compute_stage(s & 1);
  }
  // Peeled final stage.
  __syncthreads();
  asm volatile("s_wait_asynccnt 0x0" ::: "memory");
  __syncthreads();
  compute_stage(s & 1);

  // Epilogue: C/D layout -> element r: M = r + (lane<16?0:8), N = lane&15
  const int moff = (lane < 16) ? 0 : 8;
  const int ncol = lane & 15;
#pragma unroll
  for (int j = 0; j < 2; ++j) {
    const long gn = bn0 + wn * 32 + j * 16 + ncol;
    if (gn >= N) continue;
    const float bv = bias ? bias[gn] : 0.0f;
#pragma unroll
    for (int i = 0; i < MI; ++i) {
#pragma unroll
      for (int r = 0; r < 8; ++r) {
        const long gm = bm0 + wm * (TM / 2) + i * 16 + moff + r;
        if (gm < M) {
          float v = acc[i][j][r] + bv;
          if (relu) v = fmaxf(v, 0.0f);
          C[gm * (long)N + gn] = v;
        }
      }
    }
  }
}

// ---------------------------------------------------------------------------
// im2col (f32 src -> f16 patch matrix [Mpad x Kpad], zero padded).
// K ordering is (cin, ky, kx) matching OIHW weight flattening.
// src layout: nchw!=0 -> NCHW (network input), else NHWC (GEMM outputs).
// ---------------------------------------------------------------------------
__global__ void im2col_f16(const float* __restrict__ src, _Float16* __restrict__ dst,
                           int B, int Cin, int H, int W, int kh, int kw,
                           int stride, int pad, int OH, int OW,
                           int Kreal, int Kpad, long Mreal, long Mpad, int nchw)
{
  long idx = (long)blockIdx.x * blockDim.x + threadIdx.x;
  const long total = Mpad * (long)Kpad;
  if (idx >= total) return;
  const int  k = (int)(idx % Kpad);
  const long m = idx / Kpad;
  _Float16 v = (_Float16)0.0f;
  if (m < Mreal && k < Kreal) {
    const int ow = (int)(m % OW);
    long t = m / OW;
    const int oh = (int)(t % OH);
    const int b  = (int)(t / OH);
    const int kx = k % kw;
    const int t2 = k / kw;
    const int ky = t2 % kh;
    const int c  = t2 / kh;
    const int iy = oh * stride - pad + ky;
    const int ix = ow * stride - pad + kx;
    if (iy >= 0 && iy < H && ix >= 0 && ix < W) {
      float f = nchw ? src[((long)(b * Cin + c) * H + iy) * W + ix]
                     : src[(((long)b * H + iy) * W + ix) * Cin + c];
      v = (_Float16)f;
    }
  }
  dst[idx] = v;
}

// ---------------------------------------------------------------------------
// 3x3 stride-2 VALID max pool on NHWC f32
// ---------------------------------------------------------------------------
__global__ void maxpool3s2_nhwc(const float* __restrict__ src, float* __restrict__ dst,
                                int B, int H, int W, int C, int OH, int OW)
{
  long idx = (long)blockIdx.x * blockDim.x + threadIdx.x;
  const long total = (long)B * OH * OW * C;
  if (idx >= total) return;
  const int c = (int)(idx % C);
  long t = idx / C;
  const int ow = (int)(t % OW); t /= OW;
  const int oh = (int)(t % OH);
  const int b  = (int)(t / OH);
  const int y0 = oh * 2, x0 = ow * 2;
  float m = -3.402823466e+38f;
#pragma unroll
  for (int dy = 0; dy < 3; ++dy)
#pragma unroll
    for (int dx = 0; dx < 3; ++dx)
      m = fmaxf(m, src[(((long)b * H + (y0 + dy)) * W + (x0 + dx)) * C + c]);
  dst[idx] = m;
}

// ---------------------------------------------------------------------------
// ROI adaptive max pool: feat NHWC [B,H,W,C] f32 -> dst f16 [R, C*6*6]
// (flattened (c,i,j) exactly like pooled.reshape(R,-1) in the reference)
// ---------------------------------------------------------------------------
__global__ void roi_pool_f16(const float* __restrict__ feat, const float* __restrict__ rois,
                             const int* __restrict__ ridx, _Float16* __restrict__ dst,
                             int R, int C, int H, int W)
{
  long idx = (long)blockIdx.x * blockDim.x + threadIdx.x;
  const long total = (long)R * C * ROUT * ROUT;
  if (idx >= total) return;
  const int j = (int)(idx % ROUT);
  long t = idx / ROUT;
  const int i = (int)(t % ROUT); t /= ROUT;
  const int c = (int)(t % C);
  const int r = (int)(t / C);

  const float rx1 = rois[r * 4 + 0], ry1 = rois[r * 4 + 1];
  const float rx2 = rois[r * 4 + 2], ry2 = rois[r * 4 + 3];
  const int x1 = (int)floorf(rx1 * W);
  const int x2 = (int)ceilf (rx2 * W);
  const int y1 = (int)floorf(ry1 * H);
  const int y2 = (int)ceilf (ry2 * H);
  const int wc = x2 - x1, hc = y2 - y1;
  int cs = x1 + (j * wc) / ROUT;
  int ce = x1 + ((j + 1) * wc + ROUT - 1) / ROUT;
  int rs = y1 + (i * hc) / ROUT;
  int re = y1 + ((i + 1) * hc + ROUT - 1) / ROUT;
  cs = cs < 0 ? 0 : cs;  rs = rs < 0 ? 0 : rs;
  ce = ce > W ? W : ce;  re = re > H ? H : re;

  const int b = ridx[r];
  float m = -3.402823466e+38f;
  for (int y = rs; y < re; ++y)
    for (int x = cs; x < ce; ++x)
      m = fmaxf(m, feat[(((long)b * H + y) * W + x) * C + c]);

  dst[((long)r * C + c) * (ROUT * ROUT) + i * ROUT + j] = (_Float16)m;
}

// ---------------------------------------------------------------------------
// f32 weight [N,K] -> zero-padded f16 [Npad,Kpad]
// ---------------------------------------------------------------------------
__global__ void cvt_pad_w(const float* __restrict__ src, _Float16* __restrict__ dst,
                          int N, int K, int Npad, int Kpad)
{
  long idx = (long)blockIdx.x * blockDim.x + threadIdx.x;
  const long total = (long)Npad * Kpad;
  if (idx >= total) return;
  const int k = (int)(idx % Kpad);
  const int n = (int)(idx / Kpad);
  _Float16 v = (_Float16)0.0f;
  if (n < N && k < K) v = (_Float16)src[(long)n * K + k];
  dst[idx] = v;
}

// f32 -> f16 elementwise
__global__ void cvt_f32_f16(const float* __restrict__ src, _Float16* __restrict__ dst, long n)
{
  long idx = (long)blockIdx.x * blockDim.x + threadIdx.x;
  if (idx < n) dst[idx] = (_Float16)src[idx];
}

// ---------------------------------------------------------------------------
// Host side
// ---------------------------------------------------------------------------
static inline long cdivl(long a, long b) { return (a + b - 1) / b; }

extern "C" void kernel_launch(void* const* d_in, const int* in_sizes, int n_in,
                              void* d_out, int out_size, void* d_ws, size_t ws_size,
                              hipStream_t stream) {
  (void)in_sizes; (void)n_in; (void)out_size; (void)ws_size;

  const float* x    = (const float*)d_in[0];
  const float* rois = (const float*)d_in[1];
  const int*   ridx = (const int*)  d_in[2];
  const float* w1 = (const float*)d_in[3];  const float* b1 = (const float*)d_in[4];
  const float* w2 = (const float*)d_in[5];  const float* b2 = (const float*)d_in[6];
  const float* w3 = (const float*)d_in[7];  const float* b3 = (const float*)d_in[8];
  const float* w4 = (const float*)d_in[9];  const float* b4 = (const float*)d_in[10];
  const float* w5 = (const float*)d_in[11]; const float* b5 = (const float*)d_in[12];
  const float* fw1 = (const float*)d_in[13]; const float* fb1 = (const float*)d_in[14];
  const float* fw2 = (const float*)d_in[15]; const float* fb2 = (const float*)d_in[16];
  const float* cw  = (const float*)d_in[17]; const float* cb  = (const float*)d_in[18];
  const float* bw  = (const float*)d_in[19]; const float* bb  = (const float*)d_in[20];
  float* out = (float*)d_out;

  // ---- shapes -------------------------------------------------------------
  const int B = 8, R = 512;
  const int H1 = 111, M1 = B * H1 * H1, K1 = 363, K1p = 384, N1 = 64,  N1p = 128;
  const long M1p = cdivl(M1, 128) * 128;
  const int P1 = 55;
  const int M2 = B * P1 * P1, K2 = 1600, N2 = 192, N2p = 256;
  const long M2p = cdivl(M2, 128) * 128;
  const int P2 = 27;
  const int M3 = B * P2 * P2;
  const long M3p = cdivl(M3, 128) * 128;
  const int K3 = 1728, N3 = 384;
  const int K4 = 3456, N4 = 256;
  const int K5 = 2304, N5 = 256;
  const int KF1 = 9216, NF = 4096;
  const int NC = 21, NB = 84, NHp = 128;

  // ---- workspace layout ---------------------------------------------------
  char* ws = (char*)d_ws;
  size_t off = 0;
  auto alloc = [&](size_t bytes) { size_t p = off; off += (bytes + 255) & ~(size_t)255; return p; };

  _Float16* w1c = (_Float16*)(ws + alloc((size_t)N1p * K1p * 2));
  _Float16* w2c = (_Float16*)(ws + alloc((size_t)N2p * K2  * 2));
  _Float16* w3c = (_Float16*)(ws + alloc((size_t)N3  * K3  * 2));
  _Float16* w4c = (_Float16*)(ws + alloc((size_t)N4  * K4  * 2));
  _Float16* w5c = (_Float16*)(ws + alloc((size_t)N5  * K5  * 2));
  _Float16* f1c = (_Float16*)(ws + alloc((size_t)NF  * KF1 * 2));
  _Float16* f2c = (_Float16*)(ws + alloc((size_t)NF  * NF  * 2));
  _Float16* cwc = (_Float16*)(ws + alloc((size_t)NHp * NF  * 2));
  _Float16* bwc = (_Float16*)(ws + alloc((size_t)NHp * NF  * 2));

  const size_t buf0_bytes = (size_t)M2p * K2 * 2;          // largest im2col buffer
  const size_t buf1_bytes = (size_t)M1 * N1 * 4;           // largest GEMM output
  const size_t buf2_bytes = (size_t)B * P1 * P1 * N1 * 4;  // pool buffer
  _Float16* bufA = (_Float16*)(ws + alloc(buf0_bytes));
  float*    bufC = (float*)   (ws + alloc(buf1_bytes));
  float*    bufP = (float*)   (ws + alloc(buf2_bytes));

  const int T = 256;
  auto padw = [&](const float* s, _Float16* d, int N, int K, int Np, int Kp) {
    long tot = (long)Np * Kp;
    cvt_pad_w<<<dim3((unsigned)cdivl(tot, T)), T, 0, stream>>>(s, d, N, K, Np, Kp);
  };
  auto gemm128 = [&](const _Float16* A, const _Float16* W, const float* bias, float* C,
                     long M, int N, int Kp, int relu) {
    dim3 grid((unsigned)cdivl(N, BN), (unsigned)cdivl(M, 128));
    gemm_f16_wmma<128><<<grid, 256, 0, stream>>>(A, W, bias, C, (int)M, N, Kp, relu);
  };
  auto gemm64 = [&](const _Float16* A, const _Float16* W, const float* bias, float* C,
                    long M, int N, int Kp, int relu) {
    dim3 grid((unsigned)cdivl(N, BN), (unsigned)cdivl(M, 64));
    gemm_f16_wmma<64><<<grid, 256, 0, stream>>>(A, W, bias, C, (int)M, N, Kp, relu);
  };
  auto im2col = [&](const float* s, _Float16* d, int Cin, int H, int kh, int stride,
                    int pad, int OH, int Kr, int Kp, long Mr, long Mp, int nchw) {
    long tot = Mp * (long)Kp;
    im2col_f16<<<dim3((unsigned)cdivl(tot, T)), T, 0, stream>>>(
        s, d, B, Cin, H, H, kh, kh, stride, pad, OH, OH, Kr, Kp, Mr, Mp, nchw);
  };

  // ---- weight conversion (f32 -> padded f16) ------------------------------
  padw(w1, w1c, N1, K1, N1p, K1p);
  padw(w2, w2c, N2, K2, N2p, K2);
  padw(w3, w3c, N3, K3, N3, K3);
  padw(w4, w4c, N4, K4, N4, K4);
  padw(w5, w5c, N5, K5, N5, K5);
  padw(fw1, f1c, NF, KF1, NF, KF1);
  padw(fw2, f2c, NF, NF, NF, NF);
  padw(cw, cwc, NC, NF, NHp, NF);
  padw(bw, bwc, NB, NF, NHp, NF);

  // ---- conv1 + pool1 ------------------------------------------------------
  im2col(x, bufA, 3, 448, 11, 4, 2, H1, K1, K1p, M1, M1p, /*nchw=*/1);
  gemm128(bufA, w1c, b1, bufC, M1, N1, K1p, 1);              // [M1, 64] NHWC
  {
    long tot = (long)B * P1 * P1 * N1;
    maxpool3s2_nhwc<<<dim3((unsigned)cdivl(tot, T)), T, 0, stream>>>(
        bufC, bufP, B, H1, H1, N1, P1, P1);
  }
  // ---- conv2 + pool2 ------------------------------------------------------
  im2col(bufP, bufA, N1, P1, 5, 1, 2, P1, K2, K2, M2, M2p, 0);
  gemm128(bufA, w2c, b2, bufC, M2, N2, K2, 1);               // [M2, 192]
  {
    long tot = (long)B * P2 * P2 * N2;
    maxpool3s2_nhwc<<<dim3((unsigned)cdivl(tot, T)), T, 0, stream>>>(
        bufC, bufP, B, P1, P1, N2, P2, P2);
  }
  // ---- conv3/4/5 ----------------------------------------------------------
  im2col(bufP, bufA, N2, P2, 3, 1, 1, P2, K3, K3, M3, M3p, 0);
  gemm128(bufA, w3c, b3, bufC, M3, N3, K3, 1);               // C3 -> bufC
  im2col(bufC, bufA, N3, P2, 3, 1, 1, P2, K4, K4, M3, M3p, 0);
  gemm128(bufA, w4c, b4, bufP, M3, N4, K4, 1);               // C4 -> bufP
  im2col(bufP, bufA, N4, P2, 3, 1, 1, P2, K5, K5, M3, M3p, 0);
  gemm128(bufA, w5c, b5, bufC, M3, N5, K5, 1);               // C5 -> bufC [8,27,27,256]

  // ---- ROI pool -> f16 [512, 9216] ----------------------------------------
  {
    long tot = (long)R * N5 * ROUT * ROUT;
    roi_pool_f16<<<dim3((unsigned)cdivl(tot, T)), T, 0, stream>>>(
        bufC, rois, ridx, bufA, R, N5, P2, P2);
  }

  // ---- FC stack (TM=64 for occupancy: M=512 -> 8x32=256 blocks) -----------
  gemm64(bufA, f1c, fb1, bufC, R, NF, KF1, 1);
  cvt_f32_f16<<<dim3((unsigned)cdivl((long)R * NF, T)), T, 0, stream>>>(bufC, bufA, (long)R * NF);
  gemm64(bufA, f2c, fb2, bufC, R, NF, NF, 1);
  cvt_f32_f16<<<dim3((unsigned)cdivl((long)R * NF, T)), T, 0, stream>>>(bufC, bufA, (long)R * NF);

  // heads write straight into d_out: cls [512,21] then bbox [512,84]
  gemm64(bufA, cwc, cb, out, R, NC, NF, 0);
  gemm64(bufA, bwc, bb, out + (long)R * NC, R, NB, NF, 0);
}
